// StyleCrossAttention_43241730736704
// MI455X (gfx1250) — matrix operand order repaired and checked by reading, not compiled, not run on hardware
//
#include <hip/hip_runtime.h>

typedef __attribute__((ext_vector_type(16))) __bf16 v16bf;
typedef __attribute__((ext_vector_type(8)))  __bf16 v8bf;
typedef __attribute__((ext_vector_type(8)))  float  v8f;
typedef __attribute__((ext_vector_type(4)))  float  v4f;
typedef __attribute__((ext_vector_type(4)))  int    v4i;

#define B_    16
#define C_    512
#define T_    4096
#define D_    256
#define TT    128
#define NTOK  64          // 50 real tokens padded to 64
#define SCALE 0.0625f     // 256^-0.5

// ---- LDS geometry (halves). Row strides chosen for 16B-aligned b128 frag loads.
#define RSQ 264           // q_lds rows: 256 d + 8 pad
#define RSV 64            // vT rows: 64 tok
#define RSK 264           // kt rows: 256 d + 8 pad
#define RSP 72            // p rows: 64 tok + 8 pad
#define RSA 40            // x-chunk rows: 32 c + 8 pad
#define RSW 40            // weight-chunk rows: 32 k + 8 pad
#define OFF_Q   0                       // 128*264*2 = 67584
#define OFF_VT  67584                   // 256*64*2  = 32768
#define OFF_R3  100352                  // 61440 time-shared region
#define SMEM_BYTES 161792               // <= 2 workgroups per 320KB WGP

// ---- workspace layout (bytes)
#define WS_WQ  0          // bf16 Wq [256][512]           = 262144 B
#define WS_WO  262144     // bf16 Wo [512][256]           = 262144 B
#define WS_KT  524288     // bf16 tanh(K) [64][256]       = 32768 B
#define WS_VT  557056     // bf16 V^T [16][256][64]       = 524288 B  (total 1081344 B)

#if defined(__has_builtin)
#if __has_builtin(__builtin_amdgcn_global_load_async_to_lds_b128)
#define ASYNC_LDS 1
#endif
#if __has_builtin(__builtin_amdgcn_sched_group_barrier)
#define HAVE_SGB 1
#endif
#endif

#ifdef HAVE_SGB
#define SGB(mask, cnt) __builtin_amdgcn_sched_group_barrier((mask), (cnt), 0)
#else
#define SGB(mask, cnt)
#endif
// masks: 0x008 = MFMA/WMMA, 0x100 = DS read

__device__ __forceinline__ void cp16(void* lds, const void* g) {
#ifdef ASYNC_LDS
  __builtin_amdgcn_global_load_async_to_lds_b128(
      (__attribute__((address_space(1))) v4i*)g,
      (__attribute__((address_space(3))) v4i*)lds, 0, 0);
#else
  *(v8bf*)lds = *(const v8bf*)g;
#endif
}

__device__ __forceinline__ void wait_async() {
#ifdef ASYNC_LDS
#if __has_builtin(__builtin_amdgcn_s_wait_asynccnt)
  __builtin_amdgcn_s_wait_asynccnt(0);
#else
  asm volatile("s_wait_asynccnt 0" ::: "memory");
#endif
#endif
}

__device__ __forceinline__ v16bf ld2(const __bf16* p0, const __bf16* p1) {
  union { v16bf v; v8bf h[2]; } f;
  f.h[0] = *(const v8bf*)p0;   // halves 0..7
  f.h[1] = *(const v8bf*)p1;   // halves 8..15
  return f.v;
}

__device__ __forceinline__ v8f wmma_bf16(v16bf a, v16bf b, v8f c) {
  return __builtin_amdgcn_wmma_f32_16x16x32_bf16(false, a, false, b, (short)0, c,
                                                 false, false);
}

// ============================================================================
// Prep: bf16-convert Wq/Wo; compute tanh(key_emb@Wk^T+bk) padded [64][256];
//       compute V^T = (style@Wv^T+bv)^T padded [B][256][64].
// ============================================================================
__global__ __launch_bounds__(256) void prep_kernel(
    const float* __restrict__ Wq, const float* __restrict__ Wo,
    const float* __restrict__ key_emb, const float* __restrict__ Wk,
    const float* __restrict__ bk, const float* __restrict__ style,
    const float* __restrict__ Wv, const float* __restrict__ bv,
    __bf16* __restrict__ wq_bf, __bf16* __restrict__ wo_bf,
    __bf16* __restrict__ kt_bf, __bf16* __restrict__ vT_bf)
{
  int idx = blockIdx.x * 256 + threadIdx.x;
  if (idx < 131072) { wq_bf[idx] = (__bf16)Wq[idx]; return; }
  idx -= 131072;
  if (idx < 131072) { wo_bf[idx] = (__bf16)Wo[idx]; return; }
  idx -= 131072;
  if (idx < NTOK * D_) {                       // tanh(K)
    const int tok = idx >> 8, d = idx & 255;
    float v = 0.f;
    if (tok < 50) {
      float acc = bk[d];
      const float* ke = key_emb + tok * D_;
      const float* wk = Wk + d * D_;
      for (int s = 0; s < D_; ++s) acc = fmaf(ke[s], wk[s], acc);
      v = tanhf(acc);
    }
    kt_bf[tok * D_ + d] = (__bf16)v;
    return;
  }
  idx -= NTOK * D_;
  if (idx < B_ * D_ * NTOK) {                  // V^T
    const int b = idx >> 14;
    const int r = idx & 16383;
    const int d = r >> 6, tok = r & 63;
    float v = 0.f;
    if (tok < 50) {
      float acc = bv[d];
      const float* st = style + (size_t)(b * 50 + tok) * D_;
      const float* wv = Wv + d * D_;
      for (int s = 0; s < D_; ++s) acc = fmaf(st[s], wv[s], acc);
      v = acc;
    }
    vT_bf[(size_t)b * (D_ * NTOK) + d * NTOK + tok] = (__bf16)v;
  }
}

// ============================================================================
// Fused main kernel: one workgroup = one (batch, 128-row t-tile); 8 waves.
// ============================================================================
__global__ __launch_bounds__(256) void attn_main(
    const float* __restrict__ x, const float* __restrict__ bq,
    const float* __restrict__ bo,
    const __bf16* __restrict__ wq_bf, const __bf16* __restrict__ wo_bf,
    const __bf16* __restrict__ kt_bf, const __bf16* __restrict__ vT_bf,
    float* __restrict__ out)
{
  __shared__ __align__(128) unsigned char smem[SMEM_BYTES];
  __bf16* q_lds   = (__bf16*)(smem + OFF_Q);            // [128][RSQ]
  __bf16* vt_lds  = (__bf16*)(smem + OFF_VT);           // [256][RSV]
  __bf16* a_tile  = (__bf16*)(smem + OFF_R3);           // [2][128][RSA]
  __bf16* w_tile  = (__bf16*)(smem + OFF_R3 + 20480);   // [2][256][RSW]
  __bf16* kt_lds  = (__bf16*)(smem + OFF_R3);           // [64][RSK]   (after GEMM1)
  __bf16* p_lds   = (__bf16*)(smem + OFF_R3);           // [2][128][RSP] (after D)
  __bf16* wo_tile = (__bf16*)(smem + OFF_R3);           // [2][256][RSW] (phase H)

  const int tid  = threadIdx.x;
  const int lane = tid & 31;
  const int w    = tid >> 5;
  const int ln   = lane & 15;
  const int g    = lane >> 4;
  const int b    = blockIdx.x >> 5;
  const int t0   = (blockIdx.x & 31) * TT;
  const float* xb   = x   + (size_t)b * C_ * T_;
  float*       outb = out + (size_t)b * C_ * T_;
  const __bf16* vTb = vT_bf + (size_t)b * (D_ * NTOK);

  const v8f vzero = {0.f, 0.f, 0.f, 0.f, 0.f, 0.f, 0.f, 0.f};

  // ---- Phase A: async-copy V^T (32 KB) ----
  for (int i = tid; i < (D_ * NTOK) / 8; i += 256)
    cp16(vt_lds + i * 8, vTb + i * 8);

  // ---- GEMM1 staging: x chunk [32c x 128t] -> a_tile[t][c]; Wq chunk rows ----
  auto stage1 = [&](int kc, int buf) {
    __bf16* at = a_tile + buf * (TT * RSA);
    __bf16* wt = w_tile + buf * (D_ * RSW);
    {
      const int cl = tid >> 3;               // 0..31 channel in chunk
      const int ts = (tid & 7) * 16;         // 16 contiguous t per thread
      const float* src = xb + (size_t)(kc * 32 + cl) * T_ + t0 + ts;
      if (kc + 1 < 16) __builtin_prefetch(src + 32 * T_, 0, 3);
#pragma unroll
      for (int i = 0; i < 16; i += 4) {
        v4f v = *(const v4f*)(src + i);
        at[(ts + i + 0) * RSA + cl] = (__bf16)v.x;
        at[(ts + i + 1) * RSA + cl] = (__bf16)v.y;
        at[(ts + i + 2) * RSA + cl] = (__bf16)v.z;
        at[(ts + i + 3) * RSA + cl] = (__bf16)v.w;
      }
    }
    for (int i = tid; i < 1024; i += 256) {  // 256 rows x 4 x 16B
      const int n = i >> 2, cc = i & 3;
      cp16(wt + n * RSW + cc * 8, wq_bf + (size_t)n * C_ + kc * 32 + cc * 8);
    }
  };

  // ---- Phase B: GEMM1  q[128t x 256d] = x^T * Wq^T, K=512 in 16 chunks ----
  v8f qa[16];
#pragma unroll
  for (int j = 0; j < 16; ++j) qa[j] = vzero;

  stage1(0, 0);
  wait_async();
  __syncthreads();

  const int m = 16 * w + ln;  // A-fragment row (t) for this lane

  for (int kc = 0; kc < 16; ++kc) {
    if (kc < 15) stage1(kc + 1, (kc + 1) & 1);
    const __bf16* at = a_tile + (kc & 1) * (TT * RSA);
    const __bf16* wt = w_tile + (kc & 1) * (D_ * RSW);
    v16bf A = ld2(&at[m * RSA + 8 * g], &at[m * RSA + 16 + 8 * g]);
    v16bf Bq0, Bq1;
    {
      const __bf16* bp0 = &wt[(ln)*RSW + 8 * g];
      const __bf16* bp1 = &wt[(16 + ln) * RSW + 8 * g];
      Bq0 = ld2(bp0, bp0 + 16);
      Bq1 = ld2(bp1, bp1 + 16);
    }
#pragma unroll
    for (int j = 0; j < 16; ++j) {
      v16bf Bcur = (j & 1) ? Bq1 : Bq0;
      v16bf Bnew = Bcur;
      if (j + 2 < 16) {
        const __bf16* bp = &wt[(16 * (j + 2) + ln) * RSW + 8 * g];
        Bnew = ld2(bp, bp + 16);
      }
      qa[j] = wmma_bf16(A, Bcur, qa[j]);
      if (j & 1) Bq1 = Bnew; else Bq0 = Bnew;
    }
    // enforce: 6 DS reads up front (A + 2 B-frags), then {2 DS, 1 WMMA} x14,
    // then 2 trailing WMMAs  (totals: 34 DS reads, 16 WMMAs)
    SGB(0x100, 6);
#pragma unroll
    for (int i = 0; i < 14; ++i) { SGB(0x100, 2); SGB(0x008, 1); }
    SGB(0x008, 2);
    wait_async();
    __syncthreads();
  }

  // ---- copy tanh(K) into LDS (reuses GEMM1 staging region) ----
  for (int i = tid; i < 64 * 32; i += 256)
    cp16(kt_lds + (i >> 5) * RSK + (i & 31) * 8,
         kt_bf + (i >> 5) * D_ + (i & 31) * 8);

  // ---- Phase C: +bq, spill q as bf16 to q_lds in A-friendly [t][d] layout ----
#pragma unroll
  for (int j = 0; j < 16; ++j) {
    const float bqv = bq[16 * j + ln];
    const int mr = 16 * w + 8 * g;
#pragma unroll
    for (int r = 0; r < 8; ++r)
      q_lds[(mr + r) * RSQ + 16 * j + ln] = (__bf16)(qa[j][r] + bqv);
  }
  wait_async();
  __syncthreads();

  // ---- Phase D: logits L[h][t][tok] = q_h * tanh(K_h)^T ----
  v8f lac[2][4];
#pragma unroll
  for (int h = 0; h < 2; ++h)
#pragma unroll
    for (int j = 0; j < 4; ++j) lac[h][j] = vzero;

#pragma unroll
  for (int h = 0; h < 2; ++h) {
#pragma unroll
    for (int kk = 0; kk < 4; ++kk) {
      const int kd = 128 * h + 32 * kk;
      v16bf A = ld2(&q_lds[m * RSQ + kd + 8 * g],
                    &q_lds[m * RSQ + kd + 16 + 8 * g]);
      const __bf16* bp0 = &kt_lds[(ln)*RSK + kd + 8 * g];
      v16bf Bcur = ld2(bp0, bp0 + 16);
#pragma unroll
      for (int j = 0; j < 4; ++j) {
        v16bf Bnew = Bcur;
        if (j + 1 < 4) {
          const __bf16* bp = &kt_lds[(16 * (j + 1) + ln) * RSK + kd + 8 * g];
          Bnew = ld2(bp, bp + 16);
        }
        lac[h][j] = wmma_bf16(A, Bcur, lac[h][j]);
        Bcur = Bnew;
      }
      // per step: 10 DS reads, 4 WMMAs
      SGB(0x100, 4);
#pragma unroll
      for (int i = 0; i < 3; ++i) { SGB(0x100, 2); SGB(0x008, 1); }
      SGB(0x008, 1);
    }
  }
  __syncthreads();  // kt reads complete before p overwrites the region

  // ---- Phase E: masked softmax over the 50 real tokens ----
#pragma unroll
  for (int h = 0; h < 2; ++h) {
#pragma unroll
    for (int r = 0; r < 8; ++r) {
      float v0 = lac[h][0][r] * SCALE;
      float v1 = lac[h][1][r] * SCALE;
      float v2 = lac[h][2][r] * SCALE;
      float v3 = (ln < 2) ? lac[h][3][r] * SCALE : -1e30f;  // tok 48..63 -> mask >=50
      float mx = fmaxf(fmaxf(v0, v1), fmaxf(v2, v3));
#pragma unroll
      for (int s = 8; s >= 1; s >>= 1) mx = fmaxf(mx, __shfl_xor(mx, s, 16));
      float e0 = __expf(v0 - mx), e1 = __expf(v1 - mx), e2 = __expf(v2 - mx);
      float e3 = (ln < 2) ? __expf(v3 - mx) : 0.0f;
      float sm = e0 + e1 + e2 + e3;
#pragma unroll
      for (int s = 8; s >= 1; s >>= 1) sm += __shfl_xor(sm, s, 16);
      const float inv = 1.0f / sm;
      const int pr = (h * TT + 16 * w + 8 * g + r) * RSP + ln;
      p_lds[pr + 0]  = (__bf16)(e0 * inv);
      p_lds[pr + 16] = (__bf16)(e1 * inv);
      p_lds[pr + 32] = (__bf16)(e2 * inv);
      p_lds[pr + 48] = (__bf16)(e3 * inv);
    }
  }
  __syncthreads();

  // ---- Phase F: out_head[t][d] = P_h * V (K=64 tokens per head) ----
  v8f oa[16];
#pragma unroll
  for (int j = 0; j < 16; ++j) oa[j] = vzero;
#pragma unroll
  for (int kk = 0; kk < 2; ++kk) {
    v16bf A0 = ld2(&p_lds[(0 * TT + m) * RSP + 32 * kk + 8 * g],
                   &p_lds[(0 * TT + m) * RSP + 32 * kk + 16 + 8 * g]);
    v16bf A1 = ld2(&p_lds[(1 * TT + m) * RSP + 32 * kk + 8 * g],
                   &p_lds[(1 * TT + m) * RSP + 32 * kk + 16 + 8 * g]);
    v16bf Bq0, Bq1;
    {
      const __bf16* bp0 = &vt_lds[(ln)*RSV + 32 * kk + 8 * g];
      const __bf16* bp1 = &vt_lds[(16 + ln) * RSV + 32 * kk + 8 * g];
      Bq0 = ld2(bp0, bp0 + 16);
      Bq1 = ld2(bp1, bp1 + 16);
    }
#pragma unroll
    for (int jd = 0; jd < 16; ++jd) {
      v16bf Bcur = (jd & 1) ? Bq1 : Bq0;
      v16bf Bnew = Bcur;
      if (jd + 2 < 16) {
        const __bf16* bp = &vt_lds[(16 * (jd + 2) + ln) * RSV + 32 * kk + 8 * g];
        Bnew = ld2(bp, bp + 16);
      }
      oa[jd] = wmma_bf16(jd < 8 ? A0 : A1, Bcur, oa[jd]);
      if (jd & 1) Bq1 = Bnew; else Bq0 = Bnew;
    }
    // per kk: 36 DS reads, 16 WMMAs
    SGB(0x100, 8);
#pragma unroll
    for (int i = 0; i < 14; ++i) { SGB(0x100, 2); SGB(0x008, 1); }
    SGB(0x008, 2);
  }

  // ---- Phase G: spill out_head as bf16 into q_lds ([t][d], reused) ----
#pragma unroll
  for (int j = 0; j < 16; ++j) {
    const int mr = 16 * w + 8 * g;
#pragma unroll
    for (int r = 0; r < 8; ++r)
      q_lds[(mr + r) * RSQ + 16 * j + ln] = (__bf16)(oa[j][r]);
  }
  __syncthreads();

  // ---- Phase H: y^T[c][t] = Wo * out_head^T ; + bo + residual, coalesced ----
  auto stage4 = [&](int cbase, int kd, int buf) {
    __bf16* wt = wo_tile + buf * (256 * RSW);
    for (int i = tid; i < 1024; i += 256) {
      const int rr = i >> 2, cc = i & 3;
      cp16(wt + rr * RSW + cc * 8,
           wo_bf + (size_t)(cbase + rr) * D_ + kd * 32 + cc * 8);
    }
  };

  for (int half = 0; half < 2; ++half) {
    const int cbase = 256 * half;
    v8f ya[2][8];
#pragma unroll
    for (int mt = 0; mt < 2; ++mt)
#pragma unroll
      for (int jn = 0; jn < 8; ++jn) ya[mt][jn] = vzero;

    stage4(cbase, 0, 0);
    wait_async();
    __syncthreads();

    for (int kd = 0; kd < 8; ++kd) {
      if (kd < 7) stage4(cbase, kd + 1, (kd + 1) & 1);
      const __bf16* wt = wo_tile + (kd & 1) * (256 * RSW);
      const int mr0 = 32 * w + ln;
      const int mr1 = mr0 + 16;
      v16bf A0 = ld2(&wt[mr0 * RSW + 8 * g], &wt[mr0 * RSW + 16 + 8 * g]);
      v16bf A1 = ld2(&wt[mr1 * RSW + 8 * g], &wt[mr1 * RSW + 16 + 8 * g]);
      v16bf Bq0, Bq1;
      {
        const __bf16* bp0 = &q_lds[(ln)*RSQ + 32 * kd + 8 * g];
        const __bf16* bp1 = &q_lds[(16 + ln) * RSQ + 32 * kd + 8 * g];
        Bq0 = ld2(bp0, bp0 + 16);
        Bq1 = ld2(bp1, bp1 + 16);
      }
#pragma unroll
      for (int jn = 0; jn < 8; ++jn) {
        v16bf Bcur = (jn & 1) ? Bq1 : Bq0;
        v16bf Bnew = Bcur;
        if (jn + 2 < 8) {
          const __bf16* bp = &q_lds[(16 * (jn + 2) + ln) * RSQ + 32 * kd + 8 * g];
          Bnew = ld2(bp, bp + 16);
        }
        ya[0][jn] = wmma_bf16(A0, Bcur, ya[0][jn]);
        ya[1][jn] = wmma_bf16(A1, Bcur, ya[1][jn]);
        if (jn & 1) Bq1 = Bnew; else Bq0 = Bnew;
      }
      // per kd: 20 DS reads, 16 WMMAs
      SGB(0x100, 8);
#pragma unroll
      for (int i = 0; i < 6; ++i) { SGB(0x100, 2); SGB(0x008, 2); }
      SGB(0x008, 4);
      wait_async();
      __syncthreads();
    }

    // epilogue: lanes hold contiguous t -> coalesced f32 load/store; x hits L2
#pragma unroll
    for (int mt = 0; mt < 2; ++mt) {
#pragma unroll
      for (int jn = 0; jn < 8; ++jn) {
#pragma unroll
        for (int r = 0; r < 8; ++r) {
          const int c = cbase + 32 * w + 16 * mt + 8 * g + r;
          const int t = t0 + 16 * jn + ln;
          const size_t idx = (size_t)c * T_ + t;
          outb[idx] = ya[mt][jn][r] + bo[c] + xb[idx];
        }
      }
    }
  }
}

// ============================================================================
extern "C" void kernel_launch(void* const* d_in, const int* in_sizes, int n_in,
                              void* d_out, int out_size, void* d_ws, size_t ws_size,
                              hipStream_t stream) {
  const float* x       = (const float*)d_in[0];
  const float* style   = (const float*)d_in[1];
  const float* key_emb = (const float*)d_in[2];
  const float* Wq      = (const float*)d_in[3];
  const float* bq      = (const float*)d_in[4];
  const float* Wk      = (const float*)d_in[5];
  const float* bk      = (const float*)d_in[6];
  const float* Wv      = (const float*)d_in[7];
  const float* bv      = (const float*)d_in[8];
  const float* Wo      = (const float*)d_in[9];
  const float* bo      = (const float*)d_in[10];
  float* out = (float*)d_out;

  char* ws = (char*)d_ws;   // needs >= 1,081,344 bytes
  __bf16* wq_bf = (__bf16*)(ws + WS_WQ);
  __bf16* wo_bf = (__bf16*)(ws + WS_WO);
  __bf16* kt_bf = (__bf16*)(ws + WS_KT);
  __bf16* vT_bf = (__bf16*)(ws + WS_VT);

  // 131072 + 131072 + 16384 + 262144 = 540,672 threads -> 2112 blocks
  prep_kernel<<<2112, 256, 0, stream>>>(Wq, Wo, key_emb, Wk, bk, style, Wv, bv,
                                        wq_bf, wo_bf, kt_bf, vT_bf);
  attn_main<<<dim3(B_ * (T_ / TT)), dim3(256), 0, stream>>>(
      x, bq, bo, wq_bf, wo_bf, kt_bf, vT_bf, out);
}
